// TopKSkippedQwen3MoeSparseMoeBlock_71854802862408
// MI455X (gfx1250) — matrix operand order repaired and checked by reading, compiled, not run on hardware
//
#include <hip/hip_runtime.h>

// ---------------------------------------------------------------------------
// Qwen3-MoE sparse block for MI455X (gfx1250, wave32, WMMA bf16).
// Roofline: routed compute = 38.7 GFLOP; mandatory HBM traffic = 1.21 GB of
// fp32 expert weights (~52us @ 23.3 TB/s). TILE_M=64 => weights stream ~1x;
// bf16 WMMA 16x16x32 makes matrix work ~16us (memory-bound), with fp32->bf16
// weight conversion VALU co-executing with XDL WMMA.
// ---------------------------------------------------------------------------

typedef __attribute__((ext_vector_type(8)))  float  v8f;
typedef __attribute__((ext_vector_type(16))) __bf16 v16bf;
typedef __attribute__((ext_vector_type(8)))  __bf16 v8bf;

#define T_TOK   1024
#define H_DIM   2048
#define I_DIM   768
#define N_EXP   64
#define K_TOP   4
#define TILE_M  64
#define N_TILES (T_TOK / TILE_M)      // 16
#define ACT_LD  (I_DIM + 8)           // LDS row stride in halves (1552B, 16B-aligned)

#define WMMA_BF16(A, B, C) \
    __builtin_amdgcn_wmma_f32_16x16x32_bf16(false, (A), false, (B), (short)0, (C), false, false)

static __device__ __forceinline__ v8f v8f_zero() {
    v8f z = {0.f, 0.f, 0.f, 0.f, 0.f, 0.f, 0.f, 0.f};
    return z;
}

// A-operand (16-bit, 16x32): lane<16 holds K{0..7,16..23}, lane>=16 holds
// K{8..15,24..31} => two contiguous 16B chunks at p and p+16 (p pre-offset
// by 8*(lane>>4)).
static __device__ __forceinline__ v16bf load_a_bf16(const __bf16* p) {
    v8bf c0 = *(const v8bf*)(p);
    v8bf c1 = *(const v8bf*)(p + 16);
    return __builtin_shufflevector(c0, c1, 0, 1, 2, 3, 4, 5, 6, 7,
                                           8, 9, 10, 11, 12, 13, 14, 15);
}

// B-operand (16-bit, 32x16): lane = N row of W, two 8-float K chunks at
// p and p+16 (p pre-offset by 8*(lane>>4)); convert fp32 weights -> bf16 (v_cvt_pk).
static __device__ __forceinline__ v16bf load_b_f32(const float* p) {
    const float4* q = (const float4*)p;
    float4 f0 = q[0], f1 = q[1], f2 = q[4], f3 = q[5];
    float f[16] = {f0.x, f0.y, f0.z, f0.w, f1.x, f1.y, f1.z, f1.w,
                   f2.x, f2.y, f2.z, f2.w, f3.x, f3.y, f3.z, f3.w};
    v16bf r;
    #pragma unroll
    for (int s = 0; s < 16; ++s) r[s] = (__bf16)f[s];
    return r;
}

// ---------------------------------------------------------------------------
// Router (1 wave/token) + fp32->bf16 pre-conversion of activations.
// Top-4 by logit (global softmax denom cancels under top-k renorm), softmax
// over the 4 winners, atomic append into per-expert buckets.
// ---------------------------------------------------------------------------
__global__ __launch_bounds__(256) void moe_router_kernel(
    const float* __restrict__ x,        // [T, H]
    const float* __restrict__ gw,       // [E, H]
    int* __restrict__ cnt,              // [E]
    int* __restrict__ btok,             // [E, T]
    float* __restrict__ bw,             // [E, T]
    __bf16* __restrict__ xb)            // [T, H]
{
    // x -> bf16 (grid-strided, coalesced)
    for (int i = blockIdx.x * 256 + threadIdx.x; i < T_TOK * H_DIM; i += 128 * 256)
        xb[i] = (__bf16)x[i];

    const int wave = threadIdx.x >> 5;
    const int lane = threadIdx.x & 31;
    const int t    = blockIdx.x * 8 + wave;
    const float* xr = x + (size_t)t * H_DIM;

    // Each lane keeps logits for experts (lane) and (lane+32).
    float l0 = 0.f, l1 = 0.f;
    for (int e = 0; e < N_EXP; ++e) {
        const float* g = gw + (size_t)e * H_DIM;
        float s = 0.f;
        #pragma unroll 4
        for (int k = lane; k < H_DIM; k += 32) s += xr[k] * g[k];
        #pragma unroll
        for (int off = 16; off; off >>= 1) s += __shfl_xor(s, off, 32);
        if (e < 32) { if (lane == e)      l0 = s; }
        else        { if (lane == e - 32) l1 = s; }
    }

    float topv[K_TOP];
    int   topi[K_TOP];
    #pragma unroll
    for (int p = 0; p < K_TOP; ++p) {
        float m  = l0;
        int   mi = lane;
        if (l1 > m) { m = l1; mi = lane + 32; }
        #pragma unroll
        for (int off = 16; off; off >>= 1) {
            float om = __shfl_xor(m, off, 32);
            int   oi = __shfl_xor(mi, off, 32);
            if (om > m || (om == m && oi < mi)) { m = om; mi = oi; }
        }
        topv[p] = m;
        topi[p] = mi;
        if (mi == lane)      l0 = -3.4e38f;
        if (mi == lane + 32) l1 = -3.4e38f;
    }

    float mx = topv[0];
    float w[K_TOP], s = 0.f;
    #pragma unroll
    for (int p = 0; p < K_TOP; ++p) { w[p] = __expf(topv[p] - mx); s += w[p]; }
    const float inv = 1.f / s;

    if (lane == 0) {
        #pragma unroll
        for (int p = 0; p < K_TOP; ++p) {
            int e   = topi[p];
            int pos = atomicAdd(&cnt[e], 1);
            btok[e * T_TOK + pos] = t;
            bw[e * T_TOK + pos]   = w[p] * inv;
        }
    }
}

// ---------------------------------------------------------------------------
// Fused expert MLP: block = (expert, 64-token tile), 256 threads = 8 waves.
// Phase A: gate/up = X*W1^T (bf16 WMMA, shared A load), silu(g)*u -> LDS bf16.
// Phase B: out += w_tok * (act * W2^T), atomic scatter.
// ---------------------------------------------------------------------------
__global__ __launch_bounds__(256) void moe_expert_kernel(
    const __bf16* __restrict__ xb,      // [T, H] bf16
    const float*  __restrict__ w1,      // [E, 2I, H]
    const float*  __restrict__ w2,      // [E, H, I]
    const int*    __restrict__ cnt,     // [E]
    const int*    __restrict__ btok,    // [E, T]
    const float*  __restrict__ bw,      // [E, T]
    float* __restrict__ out)            // [T, H]
{
    const int e    = blockIdx.x;
    const int tile = blockIdx.y;
    const int nrow = cnt[e];
    const int m0   = tile * TILE_M;
    if (m0 >= nrow) return;                 // uniform per block

    extern __shared__ unsigned char smem[];
    __bf16* act  = (__bf16*)smem;                     // [TILE_M][ACT_LD]
    int*    toks = (int*)(act + TILE_M * ACT_LD);     // [TILE_M]
    float*  twt  = (float*)(toks + TILE_M);           // [TILE_M]

    int rows = nrow - m0;
    if (rows > TILE_M) rows = TILE_M;
    if (threadIdx.x < TILE_M) {
        int r = threadIdx.x;
        // pad rows duplicate a valid token but carry weight 0 -> contributes 0
        toks[r] = (r < rows) ? btok[e * T_TOK + m0 + r] : btok[e * T_TOK + m0];
        twt[r]  = (r < rows) ? bw[e * T_TOK + m0 + r] : 0.f;
    }
    __syncthreads();

    const int wave = threadIdx.x >> 5;
    const int lane = threadIdx.x & 31;
    const int half = lane >> 4;
    const int lr   = lane & 15;
    const int ka   = 8 * half;              // A chunk K offset (halves)
    const int kb   = 8 * half;              // B chunk K offset (floats)

    int aoff[4];
    #pragma unroll
    for (int s = 0; s < 4; ++s) aoff[s] = toks[s * 16 + lr] * H_DIM + ka;

    const float* W1e = w1 + (size_t)e * (2 * I_DIM) * H_DIM;
    const float* W2e = w2 + (size_t)e * H_DIM * I_DIM;

    // ---------------- Phase A: gate_up + SiLU -> LDS (bf16) ----------------
    // Wave w owns gate cols [w*96, w*96+96) and the matching up cols (+768).
    #pragma unroll 1
    for (int tt = 0; tt < 6; ++tt) {
        const int col = wave * 96 + tt * 16;
        const int bg  = (col + lr) * H_DIM + kb;            // gate W1 row
        const int bu  = (I_DIM + col + lr) * H_DIM + kb;    // up   W1 row

        v8f cg[4], cu[4];
        #pragma unroll
        for (int s = 0; s < 4; ++s) { cg[s] = v8f_zero(); cu[s] = v8f_zero(); }

        #pragma unroll 2
        for (int k = 0; k < H_DIM; k += 32) {
            __builtin_prefetch(W1e + bg + k + 1024, 0, 1);
            __builtin_prefetch(W1e + bu + k + 1024, 0, 1);
            v16bf bgv = load_b_f32(W1e + bg + k);
            v16bf buv = load_b_f32(W1e + bu + k);
            #pragma unroll
            for (int s = 0; s < 4; ++s) {
                v16bf av = load_a_bf16(xb + aoff[s] + k);
                cg[s] = WMMA_BF16(av, bgv, cg[s]);
                cu[s] = WMMA_BF16(av, buv, cu[s]);
            }
        }
        // C/D layout: (vgpr j, lane) -> row s*16 + j + 8*half, col = col + lr
        #pragma unroll
        for (int s = 0; s < 4; ++s) {
            #pragma unroll
            for (int j = 0; j < 8; ++j) {
                int row = s * 16 + j + 8 * half;
                float g = cg[s][j];
                float a = (g / (1.f + __expf(-g))) * cu[s][j];
                act[row * ACT_LD + col + lr] = (__bf16)a;
            }
        }
    }
    __syncthreads();

    // ---------------- Phase B: down proj + weighted scatter ----------------
    int la[4];
    #pragma unroll
    for (int s = 0; s < 4; ++s) la[s] = (s * 16 + lr) * ACT_LD + ka;

    #pragma unroll 1
    for (int ht = 0; ht < 16; ++ht) {
        const int h0 = wave * 256 + ht * 16;
        const int bo = (h0 + lr) * I_DIM + kb;
        v8f c[4];
        #pragma unroll
        for (int s = 0; s < 4; ++s) c[s] = v8f_zero();

        #pragma unroll 2
        for (int k = 0; k < I_DIM; k += 32) {
            __builtin_prefetch(W2e + bo + k + 768, 0, 1);
            v16bf bv = load_b_f32(W2e + bo + k);
            #pragma unroll
            for (int s = 0; s < 4; ++s) {
                v16bf av = load_a_bf16(act + la[s] + k);
                c[s] = WMMA_BF16(av, bv, c[s]);
            }
        }
        #pragma unroll
        for (int s = 0; s < 4; ++s) {
            #pragma unroll
            for (int j = 0; j < 8; ++j) {
                int m = s * 16 + j + 8 * half;
                atomicAdd(&out[(size_t)toks[m] * H_DIM + h0 + lr], twt[m] * c[s][j]);
            }
        }
    }
}

// ---------------------------------------------------------------------------
extern "C" void kernel_launch(void* const* d_in, const int* in_sizes, int n_in,
                              void* d_out, int out_size, void* d_ws, size_t ws_size,
                              hipStream_t stream) {
    const float* x  = (const float*)d_in[0];    // [1,1024,2048]
    const float* gw = (const float*)d_in[1];    // [64,2048]
    const float* w1 = (const float*)d_in[2];    // [64,1536,2048]
    const float* w2 = (const float*)d_in[3];    // [64,2048,768]
    float* out = (float*)d_out;                 // [1,1024,2048]

    int*    cnt  = (int*)d_ws;                  // [E]
    int*    btok = cnt + N_EXP;                 // [E, T]
    float*  bw   = (float*)(btok + N_EXP * T_TOK);
    __bf16* xb   = (__bf16*)(bw + N_EXP * T_TOK);   // [T, H] (16B-aligned)

    hipMemsetAsync(cnt, 0, N_EXP * sizeof(int), stream);
    hipMemsetAsync(out, 0, (size_t)T_TOK * H_DIM * sizeof(float), stream);

    moe_router_kernel<<<T_TOK / 8, 256, 0, stream>>>(x, gw, cnt, btok, bw, xb);

    size_t shmem = (size_t)TILE_M * ACT_LD * sizeof(__bf16) + TILE_M * 8;
    moe_expert_kernel<<<dim3(N_EXP, N_TILES), 256, shmem, stream>>>(
        xb, w1, w2, cnt, btok, bw, out);
}
